// MultiHeadLatentAttention_26328149524812
// MI455X (gfx1250) — compile-verified, matrix operand order checked by reading, not run on hardware
//
#include <hip/hip_runtime.h>

// ---------------------------------------------------------------------------
// MLA forward for MI455X (gfx1250, wave32, WMMA bf16 16x16x32, f32 accum)
// Async global->LDS (ASYNCcnt) double-buffered staging for GEMM + attention.
// ---------------------------------------------------------------------------

typedef __attribute__((ext_vector_type(16))) __bf16 bf16x16;
typedef __attribute__((ext_vector_type(8)))  __bf16 bf16x8;
typedef __attribute__((ext_vector_type(8)))  float  f32x8;

#define Bc   2
#define Sc   2048
#define HID  2048
#define NH   16
#define HD   128
#define LAT  512
#define MROWS (Bc*Sc)

static __device__ inline bf16x16 make_frag(const __bf16* lo) {
  // lo points at 8 contiguous bf16 (K block a), lo+16 is K block a+16.
  bf16x8 a = *(const bf16x8*)lo;
  bf16x8 b = *(const bf16x8*)(lo + 16);
  bf16x16 r;
#pragma unroll
  for (int i = 0; i < 8; ++i) { r[i] = a[i]; r[i + 8] = b[i]; }
  return r;
}

// Async DMA: global (16B per lane) -> LDS, tracked by ASYNCcnt.
static __device__ inline void async_copy_b128(const void* lds_dst, const void* gsrc) {
  unsigned lds_off = (unsigned)(unsigned long long)lds_dst;   // addr[31:0] = LDS offset
  unsigned long long ga = (unsigned long long)gsrc;
  asm volatile("global_load_async_to_lds_b128 %0, %1, off"
               :: "v"(lds_off), "v"(ga) : "memory");
}
static __device__ inline void wait_async0() {
  asm volatile("s_wait_asynccnt 0x0" ::: "memory");
}

// ---------------------------- pack / transpose -----------------------------

__global__ void pack_bf16(const float* __restrict__ in, __bf16* __restrict__ out,
                          long n) {
  long i = (long)blockIdx.x * blockDim.x + threadIdx.x;
  if (i < n) out[i] = (__bf16)in[i];
}

// in: K x N (f32, row-major)  ->  out: N x K (bf16, row-major)
__global__ void packT_bf16(const float* __restrict__ in, __bf16* __restrict__ out,
                           int K, int N) {
  long i = (long)blockIdx.x * blockDim.x + threadIdx.x;
  long total = (long)K * N;
  if (i < total) {
    int n = (int)(i % N);
    int k = (int)(i / N);
    out[(long)n * K + k] = (__bf16)in[i];
  }
}

// ------------------------------- GEMM (NT) ---------------------------------
// C(MxN) = A(MxK) * B(KxN), with B supplied transposed: BT is N x K row-major.
// Block tile 128x128, K-stage 64, double-buffered async LDS staging.
// 8 waves: 2x4 grid, each wave 64x32.

#define BM 128
#define BN 128
#define BK 64
#define KPAD 8   // LDS row stride = 72 bf16 = 144B, 16B aligned

template<bool STORE_BF16>
__global__ __launch_bounds__(256) void gemm_nt(
    const __bf16* __restrict__ A, const __bf16* __restrict__ BT,
    void* __restrict__ Cp, int M, int N, int K) {
  __shared__ __bf16 As[2][BM][BK + KPAD];
  __shared__ __bf16 Bs[2][BM][BK + KPAD];

  const int tid  = threadIdx.x;
  const int lane = tid & 31;
  const int w    = tid >> 5;
  const int hi   = lane >> 4;      // 0 or 1
  const int lm   = lane & 15;
  const int wm   = w >> 2;         // 0..1
  const int wn   = w & 3;          // 0..3
  const int bm0  = blockIdx.y * BM;
  const int bn0  = blockIdx.x * BN;

  // issue one K-stage of async copies (8 x 16B per thread)
  auto issue = [&](int buf, int k0) {
#pragma unroll
    for (int i = 0; i < 4; ++i) {
      int c  = tid + i * 256;          // 0..1023
      int r  = c >> 3;
      int cc = (c & 7) * 8;
      async_copy_b128(&As[buf][r][cc], &A [(size_t)(bm0 + r) * K + k0 + cc]);
      async_copy_b128(&Bs[buf][r][cc], &BT[(size_t)(bn0 + r) * K + k0 + cc]);
    }
  };

  f32x8 acc[4][2];
#pragma unroll
  for (int i = 0; i < 4; ++i)
#pragma unroll
    for (int j = 0; j < 2; ++j)
#pragma unroll
      for (int e = 0; e < 8; ++e) acc[i][j][e] = 0.0f;

  const int nst = K / BK;
  issue(0, 0);

  for (int s = 0; s < nst; ++s) {
    wait_async0();          // this wave's stage-s copies have landed in LDS
    __syncthreads();        // everyone's have; also fences reuse of other buffer
    if (s + 1 < nst) issue((s + 1) & 1, (s + 1) * BK);  // overlap with compute
    const int cb = s & 1;

#pragma unroll
    for (int ks = 0; ks < 2; ++ks) {
      const int kk = ks * 32;
      bf16x16 af[4], bfr[2];
#pragma unroll
      for (int nt = 0; nt < 2; ++nt)
        bfr[nt] = make_frag(&Bs[cb][wn * 32 + nt * 16 + lm][kk + hi * 8]);
#pragma unroll
      for (int mt = 0; mt < 4; ++mt)
        af[mt] = make_frag(&As[cb][wm * 64 + mt * 16 + lm][kk + hi * 8]);
#pragma unroll
      for (int mt = 0; mt < 4; ++mt)
#pragma unroll
        for (int nt = 0; nt < 2; ++nt)
          acc[mt][nt] = __builtin_amdgcn_wmma_f32_16x16x32_bf16(
              false, af[mt], false, bfr[nt], (short)0, acc[mt][nt], false, false);
    }
  }

#pragma unroll
  for (int mt = 0; mt < 4; ++mt)
#pragma unroll
    for (int nt = 0; nt < 2; ++nt)
#pragma unroll
      for (int r = 0; r < 8; ++r) {
        int row = bm0 + wm * 64 + mt * 16 + hi * 8 + r;
        int col = bn0 + wn * 32 + nt * 16 + lm;
        if (STORE_BF16)
          ((__bf16*)Cp)[(size_t)row * N + col] = (__bf16)acc[mt][nt][r];
        else
          ((float*)Cp)[(size_t)row * N + col] = acc[mt][nt][r];
      }
}

// ------------------------- RoPE + head-major relayout ----------------------
// in : (b,s,h,d) bf16   out: (b,h,s,d) bf16  (rope applied in f32)

__global__ void rope_relayout(const __bf16* __restrict__ qin,
                              const __bf16* __restrict__ kin,
                              __bf16* __restrict__ qout,
                              __bf16* __restrict__ kout) {
  long i = (long)blockIdx.x * blockDim.x + threadIdx.x;
  const long total = (long)Bc * Sc * NH * (HD / 2);
  if (i >= total) return;
  int j = (int)(i & 63);            // 0..63 (pair index)
  long t = i >> 6;
  int h = (int)(t & (NH - 1));
  t >>= 4;
  int s = (int)(t % Sc);
  int b = (int)(t / Sc);

  // inv_freq = 10000^(-2j/128)
  const float LOG_THETA = 9.210340371976184f; // ln(10000)
  float invf = __expf(-((float)j / 64.0f) * LOG_THETA);
  float fr = (float)s * invf;
  float c = __cosf(fr), sn = __sinf(fr);

  size_t ib = (((size_t)b * Sc + s) * NH + h) * HD;
  size_t ob = (((size_t)b * NH + h) * Sc + s) * HD;

  float q1 = (float)qin[ib + j], q2 = (float)qin[ib + j + 64];
  float k1 = (float)kin[ib + j], k2 = (float)kin[ib + j + 64];
  qout[ob + j]      = (__bf16)(q1 * c - q2 * sn);
  qout[ob + j + 64] = (__bf16)(q2 * c + q1 * sn);
  kout[ob + j]      = (__bf16)(k1 * c - k2 * sn);
  kout[ob + j + 64] = (__bf16)(k2 * c + k1 * sn);
}

// V relayout: (b,s,h,d) -> (b,h,d,s)   (V^T per head, so PV B-frags are NT loads)
__global__ void v_relayout(const __bf16* __restrict__ vin,
                           __bf16* __restrict__ vout) {
  long i = (long)blockIdx.x * blockDim.x + threadIdx.x;
  const long total = (long)Bc * Sc * HID;
  if (i >= total) return;
  int d = (int)(i & (HD - 1));
  long t = i >> 7;
  int h = (int)(t & (NH - 1));
  t >>= 4;
  int s = (int)(t % Sc);
  int b = (int)(t / Sc);
  vout[(((size_t)b * NH + h) * HD + d) * Sc + s] = vin[i];
}

// ------------------------------ flash attention ----------------------------
// Q,K: (b,h,s,d) bf16.  Vt: (b,h,d,s) bf16.  Out: (b,s,h*d) bf16.
// Block: 128 queries x one (b,h).  8 waves, 16 queries per wave.
// KV tiles of 32 keys, double-buffered async LDS staging.
// Scores computed transposed (S^T = K*Q^T) so the score C-layout (lane=query)
// is directly the A-fragment layout of P for P@V -> no LDS transpose.

__global__ __launch_bounds__(256) void flash_attn(
    const __bf16* __restrict__ Q, const __bf16* __restrict__ Kg,
    const __bf16* __restrict__ Vt, __bf16* __restrict__ Ob) {
  __shared__ __bf16 Ks[2][32][HD + 8];   // 32 keys x 128 d   (stride 272B)
  __shared__ __bf16 Vs[2][HD][32 + 8];   // 128 d x 32 keys   (stride 80B)

  const int qb   = blockIdx.x * 128;
  const int h    = blockIdx.y;
  const int b    = blockIdx.z;
  const int tid  = threadIdx.x;
  const int lane = tid & 31;
  const int w    = tid >> 5;
  const int hi   = lane >> 4;
  const int lm   = lane & 15;

  const size_t bh = (size_t)b * NH + h;
  const __bf16* qp = Q  + bh * (size_t)Sc * HD;
  const __bf16* kp = Kg + bh * (size_t)Sc * HD;
  const __bf16* vp = Vt + bh * (size_t)HD * Sc;

  // issue one KV tile (4 x 16B async copies per thread)
  auto issue = [&](int buf, int kt) {
#pragma unroll
    for (int i = 0; i < 2; ++i) {
      int c = tid + i * 256;                 // 0..511
      int r  = c >> 4, cc = (c & 15) * 8;    // K tile: 32 rows x 16 chunks
      async_copy_b128(&Ks[buf][r][cc], &kp[(size_t)(kt * 32 + r) * HD + cc]);
      int r2 = c >> 2, c2 = (c & 3) * 8;     // V^T tile: 128 rows x 4 chunks
      async_copy_b128(&Vs[buf][r2][c2], &vp[(size_t)r2 * Sc + kt * 32 + c2]);
    }
  };

  const int qrow = qb + w * 16 + lm;   // this lane's query (column of S^T)

  // Q fragments (B operand of S^T = K*Q^T), kept in registers the whole loop.
  bf16x16 qf[4];
#pragma unroll
  for (int ks = 0; ks < 4; ++ks)
    qf[ks] = make_frag(qp + (size_t)qrow * HD + ks * 32 + hi * 8);

  f32x8 o[8];
#pragma unroll
  for (int nt = 0; nt < 8; ++nt)
#pragma unroll
    for (int e = 0; e < 8; ++e) o[nt][e] = 0.0f;

  float m = -1e30f, l = 0.0f;
  const float scale = 0.08838834764831845f;  // 1/sqrt(128)
  const int ntiles = qb / 32 + 4;            // causal: keys up to qb+127

  issue(0, 0);

  for (int kt = 0; kt < ntiles; ++kt) {
    wait_async0();
    __syncthreads();
    if (kt + 1 < ntiles) issue((kt + 1) & 1, kt + 1);  // overlap DMA w/ compute
    const int cb = kt & 1;

    // S^T tiles: two 16(keys) x 16(queries) WMMAs, K-dim = head dim (4 steps)
    f32x8 st[2];
#pragma unroll
    for (int t = 0; t < 2; ++t) {
#pragma unroll
      for (int e = 0; e < 8; ++e) st[t][e] = 0.0f;
#pragma unroll
      for (int ks = 0; ks < 4; ++ks) {
        bf16x16 af = make_frag(&Ks[cb][t * 16 + lm][ks * 32 + hi * 8]);
        st[t] = __builtin_amdgcn_wmma_f32_16x16x32_bf16(
            false, af, false, qf[ks], (short)0, st[t], false, false);
      }
    }

    // scale + causal mask + online softmax (per-lane = per-query)
    float pv[2][8];
    float vmax = -1e30f;
#pragma unroll
    for (int t = 0; t < 2; ++t)
#pragma unroll
      for (int r = 0; r < 8; ++r) {
        int key = kt * 32 + t * 16 + hi * 8 + r;
        float v = st[t][r] * scale;
        if (key > qrow) v = -1e30f;
        pv[t][r] = v;
        vmax = fmaxf(vmax, v);
      }
    vmax = fmaxf(vmax, __shfl_xor(vmax, 16, 32));
    float mn = fmaxf(m, vmax);
    float alpha = __expf(m - mn);

    bf16x16 pf;
    float rs = 0.0f;
#pragma unroll
    for (int t = 0; t < 2; ++t)
#pragma unroll
      for (int r = 0; r < 8; ++r) {
        float p = __expf(pv[t][r] - mn);
        rs += p;
        pf[t * 8 + r] = (__bf16)p;   // C-layout of S^T == A-frag layout of P
      }
    rs += __shfl_xor(rs, 16, 32);
    l = l * alpha + rs;
    m = mn;

    // broadcast per-O-row rescale factors (O row r+8*hi -> stats lane r+8*hi)
    float ab[8];
#pragma unroll
    for (int r = 0; r < 8; ++r) ab[r] = __shfl(alpha, r + hi * 8, 32);
#pragma unroll
    for (int nt = 0; nt < 8; ++nt)
#pragma unroll
      for (int r = 0; r < 8; ++r) o[nt][r] *= ab[r];

    // O += P @ V   (B-frag of V from V^T rows in LDS)
#pragma unroll
    for (int nt = 0; nt < 8; ++nt) {
      bf16x16 bfr = make_frag(&Vs[cb][nt * 16 + lm][hi * 8]);
      o[nt] = __builtin_amdgcn_wmma_f32_16x16x32_bf16(
          false, pf, false, bfr, (short)0, o[nt], false, false);
    }
  }

  float lb[8];
#pragma unroll
  for (int r = 0; r < 8; ++r) lb[r] = __shfl(l, r + hi * 8, 32);

#pragma unroll
  for (int nt = 0; nt < 8; ++nt)
#pragma unroll
    for (int r = 0; r < 8; ++r) {
      int row = qb + w * 16 + hi * 8 + r;
      int col = h * HD + nt * 16 + lm;
      Ob[((size_t)b * Sc + row) * HID + col] = (__bf16)(o[nt][r] / lb[r]);
    }
}

// --------------------------------- launch ----------------------------------

extern "C" void kernel_launch(void* const* d_in, const int* in_sizes, int n_in,
                              void* d_out, int out_size, void* d_ws, size_t ws_size,
                              hipStream_t stream) {
  (void)in_sizes; (void)n_in; (void)out_size; (void)ws_size;
  const float* x     = (const float*)d_in[0];
  const float* Wq_d  = (const float*)d_in[1];
  const float* Wkv_d = (const float*)d_in[2];
  const float* Wq_u  = (const float*)d_in[3];
  const float* Wk_u  = (const float*)d_in[4];
  const float* Wv_u  = (const float*)d_in[5];
  const float* Wo    = (const float*)d_in[6];

  char* wp = (char*)d_ws;
  auto take = [&](size_t bytes) -> void* {
    void* p = (void*)wp;
    wp += (bytes + 255) & ~(size_t)255;
    return p;
  };

  __bf16* xb     = (__bf16*)take((size_t)MROWS * HID * 2);
  __bf16* wqdT   = (__bf16*)take((size_t)LAT * HID * 2);
  __bf16* wkvdT  = (__bf16*)take((size_t)LAT * HID * 2);
  __bf16* wquT   = (__bf16*)take((size_t)HID * LAT * 2);
  __bf16* wkuT   = (__bf16*)take((size_t)HID * LAT * 2);
  __bf16* wvuT   = (__bf16*)take((size_t)HID * LAT * 2);
  __bf16* woT    = (__bf16*)take((size_t)HID * HID * 2);
  __bf16* qlat   = (__bf16*)take((size_t)MROWS * LAT * 2);
  __bf16* kvlat  = (__bf16*)take((size_t)MROWS * LAT * 2);
  __bf16* q_bshd = (__bf16*)take((size_t)MROWS * HID * 2);
  __bf16* k_bshd = (__bf16*)take((size_t)MROWS * HID * 2);
  __bf16* v_bshd = (__bf16*)take((size_t)MROWS * HID * 2);
  __bf16* q_bhsd = (__bf16*)take((size_t)MROWS * HID * 2);
  __bf16* k_bhsd = (__bf16*)take((size_t)MROWS * HID * 2);
  __bf16* v_bhds = (__bf16*)take((size_t)MROWS * HID * 2);
  __bf16* attn   = (__bf16*)take((size_t)MROWS * HID * 2);

  const long nx = (long)MROWS * HID;
  pack_bf16<<<(nx + 255) / 256, 256, 0, stream>>>(x, xb, nx);
  packT_bf16<<<((long)HID * LAT + 255) / 256, 256, 0, stream>>>(Wq_d,  wqdT,  HID, LAT);
  packT_bf16<<<((long)HID * LAT + 255) / 256, 256, 0, stream>>>(Wkv_d, wkvdT, HID, LAT);
  packT_bf16<<<((long)LAT * HID + 255) / 256, 256, 0, stream>>>(Wq_u,  wquT,  LAT, HID);
  packT_bf16<<<((long)LAT * HID + 255) / 256, 256, 0, stream>>>(Wk_u,  wkuT,  LAT, HID);
  packT_bf16<<<((long)LAT * HID + 255) / 256, 256, 0, stream>>>(Wv_u,  wvuT,  LAT, HID);
  packT_bf16<<<((long)HID * HID + 255) / 256, 256, 0, stream>>>(Wo,    woT,   HID, HID);

  // down-projections: (4096 x 2048) @ (2048 x 512)
  gemm_nt<true><<<dim3(LAT / BN, MROWS / BM), 256, 0, stream>>>(xb, wqdT,  qlat,  MROWS, LAT, HID);
  gemm_nt<true><<<dim3(LAT / BN, MROWS / BM), 256, 0, stream>>>(xb, wkvdT, kvlat, MROWS, LAT, HID);

  // up-projections: (4096 x 512) @ (512 x 2048)
  gemm_nt<true><<<dim3(HID / BN, MROWS / BM), 256, 0, stream>>>(qlat,  wquT, q_bshd, MROWS, HID, LAT);
  gemm_nt<true><<<dim3(HID / BN, MROWS / BM), 256, 0, stream>>>(kvlat, wkuT, k_bshd, MROWS, HID, LAT);
  gemm_nt<true><<<dim3(HID / BN, MROWS / BM), 256, 0, stream>>>(kvlat, wvuT, v_bshd, MROWS, HID, LAT);

  // RoPE + relayouts
  const long nr = (long)Bc * Sc * NH * (HD / 2);
  rope_relayout<<<(nr + 255) / 256, 256, 0, stream>>>(q_bshd, k_bshd, q_bhsd, k_bhsd);
  const long nv = (long)Bc * Sc * HID;
  v_relayout<<<(nv + 255) / 256, 256, 0, stream>>>(v_bshd, v_bhds);

  // flash attention: 128 queries per block, per (b, h)
  flash_attn<<<dim3(Sc / 128, NH, Bc), 256, 0, stream>>>(q_bhsd, k_bhsd, v_bhds, attn);

  // output projection -> f32 d_out
  gemm_nt<false><<<dim3(HID / BN, MROWS / BM), 256, 0, stream>>>(attn, woT, (float*)d_out, MROWS, HID, HID);
}
